// SpatialSceneGraphConstructor_45672682225860
// MI455X (gfx1250) — compile-verified, matrix operand order checked by reading, not run on hardware
//
#include <hip/hip_runtime.h>
#include <hip/hip_bf16.h>
#include <math.h>

// ---------------------------------------------------------------------------
// SpatialSceneGraphConstructor for MI455X (gfx1250, wave32, WMMA 16x16x32 f16)
// B=128, C=768, H=W=14 (N=196), D=128, K=8
// ---------------------------------------------------------------------------

#define BB 128
#define CC 768
#define HH 14
#define NN 196      // H*W
#define DD 128
#define KK 8
#define NPAD 208    // 13 * 16, padded row count per batch for similarity GEMM

typedef __attribute__((ext_vector_type(16))) _Float16 v16h;
typedef __attribute__((ext_vector_type(8)))  float    v8f;

union FragH {
    v16h  v;
    uint4 q[2];
};

__device__ __forceinline__ v8f wmma_f16(v16h a, v16h b, v8f c) {
    return __builtin_amdgcn_wmma_f32_16x16x32_f16(
        false, a, false, b, (short)0, c, false, false);
}

__device__ __forceinline__ float gelu_exact(float x) {
    return 0.5f * x * (1.0f + erff(x * 0.70710678118654752f));
}

// A-style fragment from a row-major f16 matrix, K contiguous (ISA 7.12.2):
//   lane<16 : M=lane,    j=0..7 -> K=k0+j,   j=8..15 -> K=k0+16+(j-8)
//   lane>=16: M=lane-16, j=0..7 -> K=k0+8+j, j=8..15 -> K=k0+24+(j-8)
__device__ __forceinline__ v16h load_frag_rowmajor(const _Float16* base,
                                                   int row_stride, int r0,
                                                   int k0, int lane) {
    FragH f;
    const int m    = lane & 15;
    const int koff = k0 + ((lane & 16) ? 8 : 0);
    const _Float16* p = base + (size_t)(r0 + m) * row_stride + koff;
    f.q[0] = *(const uint4*)p;
    f.q[1] = *(const uint4*)(p + 16);
    return f.v;
}

// B-fragment from packed weights: tile = ks*8 + nt (a K-step's 8 tiles are
// contiguous 8KB). 32 lanes x 16 f16 per tile.
__device__ __forceinline__ FragH load_frag_packed(const _Float16* pack,
                                                  int tile, int lane) {
    FragH f;
    const _Float16* p = pack + ((size_t)tile * 32 + lane) * 16;
    f.q[0] = *(const uint4*)p;
    f.q[1] = *(const uint4*)(p + 16);
    return f;
}

// Async global -> LDS copy, 16B per lane (CDNA5 GLOBAL_LOAD_ASYNC_TO_LDS_B128,
// tracked by ASYNCcnt).
__device__ __forceinline__ void async_copy_b128(unsigned lds_off, const void* gaddr) {
    asm volatile("global_load_async_to_lds_b128 %0, %1, off"
                 :: "v"(lds_off), "v"(gaddr)
                 : "memory");
}
__device__ __forceinline__ void wait_async0() {
#if __has_builtin(__builtin_amdgcn_s_wait_asynccnt)
    __builtin_amdgcn_s_wait_asynccnt(0);
#else
    asm volatile("s_wait_asynccnt 0x0" ::: "memory");
#endif
}

// ---------------------------------------------------------------------------
// Kernel 0: pack weight matrices [K x 128] f32 into WMMA B-fragment order.
// tile index inside a matrix = ks*8 + nt.
// blocks: W_node 192, We1 96, We2 32, Wp2 16  -> 336
// ---------------------------------------------------------------------------
__global__ __launch_bounds__(32) void pack_weights_kernel(
    const float* __restrict__ Wn,  const float* __restrict__ We1,
    const float* __restrict__ We2, const float* __restrict__ Wp2,
    _Float16* __restrict__ pWn,  _Float16* __restrict__ pWe1,
    _Float16* __restrict__ pWe2, _Float16* __restrict__ pWp2)
{
    int t = blockIdx.x;
    const float* src;
    _Float16*    dst;
    if (t < 192)      {           src = Wn;  dst = pWn;  }
    else if (t < 288) { t -= 192; src = We1; dst = pWe1; }
    else if (t < 320) { t -= 288; src = We2; dst = pWe2; }
    else              { t -= 320; src = Wp2; dst = pWp2; }
    const int ks   = t >> 3;     // tile = ks*8 + nt
    const int nt   = t & 7;
    const int lane = threadIdx.x;
    _Float16* out = dst + ((size_t)t * 32 + lane) * 16;
    const int nloc  = nt * 16 + (lane & 15);
    const int kbase = ks * 32 + ((lane & 16) ? 8 : 0);
#pragma unroll
    for (int j = 0; j < 16; ++j) {
        int k = kbase + ((j < 8) ? j : (8 + j));
        out[j] = (_Float16)src[(size_t)k * DD + nloc];
    }
}

__global__ void zero_pad_kernel(_Float16* __restrict__ nrm) {
    int idx = blockIdx.x * blockDim.x + threadIdx.x;
    const int total = BB * (NPAD - NN) * DD;
    if (idx < total) {
        int b   = idx / ((NPAD - NN) * DD);
        int r   = idx % ((NPAD - NN) * DD);
        int row = NN + r / DD;
        int d   = r % DD;
        nrm[((size_t)b * NPAD + row) * DD + d] = (_Float16)0.0f;
    }
}

// ---------------------------------------------------------------------------
// Kernel 1: nodes = GELU(LN(tokens @ W_node + b)). 4 waves/block, 16 rows per
// wave. Batched B-fragments: 16 loads in flight per K-step, then 8 WMMAs.
// ---------------------------------------------------------------------------
__global__ __launch_bounds__(128) void node_proj_kernel(
    const float* __restrict__ feat_map, const float* __restrict__ b_node,
    const float* __restrict__ ln_g,     const float* __restrict__ ln_b,
    const _Float16* __restrict__ pWn,
    float* __restrict__ out_nodes, _Float16* __restrict__ nodes_f16,
    _Float16* __restrict__ nrm_f16)
{
    __shared__ __align__(16) float smem[4][16][DD];
    const int tid  = threadIdx.x;
    const int wave = tid >> 5;
    const int lane = tid & 31;
    float (*sm)[DD] = smem[wave];

    const int r0 = (blockIdx.x * 4 + wave) * 16;
    const int m  = lane & 15;
    const int rr = r0 + m;
    const int b  = rr / NN;
    const int n  = rr % NN;
    const float* fm = feat_map + (size_t)b * CC * NN + n;   // c-stride = NN

    v8f acc[8];
    const v8f vz = {0.f, 0.f, 0.f, 0.f, 0.f, 0.f, 0.f, 0.f};
#pragma unroll
    for (int t = 0; t < 8; ++t) acc[t] = vz;

    for (int ks = 0; ks < CC / 32; ++ks) {
        const int k0   = ks * 32;
        const int koff = k0 + ((lane & 16) ? 8 : 0);
        FragH a;
#pragma unroll
        for (int j = 0; j < 16; ++j) {
            int c  = koff + ((j < 8) ? j : (8 + j));
            a.v[j] = (_Float16)fm[(size_t)c * NN];
        }
        FragH bfr[8];
#pragma unroll
        for (int nt = 0; nt < 8; ++nt)
            bfr[nt] = load_frag_packed(pWn, ks * 8 + nt, lane);
        if (ks + 1 < CC / 32)   // pull next 8KB weight block toward the WGP
            __builtin_prefetch(pWn + (size_t)(ks + 1) * 8 * 512 + lane * 128, 0, 1);
#pragma unroll
        for (int nt = 0; nt < 8; ++nt)
            acc[nt] = wmma_f16(a.v, bfr[nt].v, acc[nt]);
    }

    {   // C/D layout: elem r -> (M = r + 8*(lane>=16), N = nt*16 + lane%16)
        const int mrow = (lane >> 4) * 8;
        const int ncol = lane & 15;
#pragma unroll
        for (int nt = 0; nt < 8; ++nt)
#pragma unroll
            for (int r = 0; r < 8; ++r)
                sm[r + mrow][nt * 16 + ncol] = acc[nt][r];
    }
    __syncthreads();

    if (lane < 16) {
        float mu = 0.f;
        for (int d = 0; d < DD; ++d) mu += sm[m][d] + b_node[d];
        mu *= (1.0f / DD);
        float var = 0.f;
        for (int d = 0; d < DD; ++d) {
            float v = sm[m][d] + b_node[d] - mu;
            var += v * v;
        }
        var *= (1.0f / DD);
        const float rstd = rsqrtf(var + 1e-5f);
        float ss = 0.f;
        for (int d = 0; d < DD; ++d) {
            float v = sm[m][d] + b_node[d];
            float y = (v - mu) * rstd * ln_g[d] + ln_b[d];
            float g = gelu_exact(y);
            sm[m][d] = g;
            ss += g * g;
            out_nodes[(size_t)rr * DD + d] = g;
        }
        const float inv = 1.0f / fmaxf(sqrtf(ss), 1e-12f);
        _Float16* nf = nodes_f16 + (size_t)rr * DD;
        _Float16* nr = nrm_f16 + ((size_t)b * NPAD + n) * DD;
        for (int d = 0; d < DD; ++d) {
            float g = sm[m][d];
            nf[d] = (_Float16)g;
            nr[d] = (_Float16)(g * inv);
        }
    }
}

// ---------------------------------------------------------------------------
// Kernel 2: sim = nrm @ nrm^T, mask, top-8 per row. 4 waves/block, 16-row
// strip per wave; 13 col tiles x 4 K-steps, B-fragments batched 7+6.
// ---------------------------------------------------------------------------
__global__ __launch_bounds__(128) void simtopk_kernel(
    const _Float16* __restrict__ nrm_f16, int* __restrict__ adj_i,
    float* __restrict__ out_adj)
{
    __shared__ __align__(16) float sim4[4][16][NPAD];
    const int tid  = threadIdx.x;
    const int wave = tid >> 5;
    const int lane = tid & 31;
    float (*sim)[NPAD] = sim4[wave];

    const int strip = blockIdx.x * 4 + wave;
    const int b     = strip / 13;
    const int n0    = (strip % 13) * 16;
    const _Float16* base = nrm_f16 + (size_t)b * NPAD * DD;

    v8f acc[13];
    const v8f vz = {0.f, 0.f, 0.f, 0.f, 0.f, 0.f, 0.f, 0.f};
#pragma unroll
    for (int t = 0; t < 13; ++t) acc[t] = vz;

    for (int ks = 0; ks < DD / 32; ++ks) {
        const int k0 = ks * 32;
        v16h a = load_frag_rowmajor(base, DD, n0, k0, lane);
        FragH bf[7];
#pragma unroll
        for (int ct = 0; ct < 7; ++ct) {
            FragH f;
            f.v = load_frag_rowmajor(base, DD, ct * 16, k0, lane);
            bf[ct] = f;
        }
#pragma unroll
        for (int ct = 0; ct < 7; ++ct)
            acc[ct] = wmma_f16(a, bf[ct].v, acc[ct]);
#pragma unroll
        for (int ct = 0; ct < 6; ++ct) {
            FragH f;
            f.v = load_frag_rowmajor(base, DD, (ct + 7) * 16, k0, lane);
            bf[ct] = f;
        }
#pragma unroll
        for (int ct = 0; ct < 6; ++ct)
            acc[ct + 7] = wmma_f16(a, bf[ct].v, acc[ct + 7]);
    }
    {
        const int mrow = (lane >> 4) * 8;
        const int ncol = lane & 15;
#pragma unroll
        for (int ct = 0; ct < 13; ++ct)
#pragma unroll
            for (int r = 0; r < 8; ++r)
                sim[r + mrow][ct * 16 + ncol] = acc[ct][r];
    }
    __syncthreads();

    if (lane < 16) {
        const int n = n0 + lane;
        if (n < NN) {
            float* row = sim[lane];
            row[n] -= 1e9f;
            for (int c = NN; c < NPAD; ++c) row[c] = -3.0e38f;
            for (int t = 0; t < KK; ++t) {
                float best = -3.4e38f;
                int   bi   = 0;
                for (int c = 0; c < NPAD; ++c)
                    if (row[c] > best) { best = row[c]; bi = c; }
                row[bi] = -3.0e38f;
                const size_t o = ((size_t)b * NN + n) * KK + t;
                adj_i[o]   = bi;
                out_adj[o] = (float)bi;
            }
        }
    }
}

// ---------------------------------------------------------------------------
// Kernel 3: edge MLPs. 4 waves/block, per-wave LDS slab in one merged shared
// block (LDS byte offsets well-defined for async copies). Node rows are
// staged global->LDS with GLOBAL_LOAD_ASYNC_TO_LDS_B128 (ASYNCcnt).
// WMMAs per wave: 16 (spatial) + 96 (GEMM1) + 32 (GEMM2) = 144.
// ---------------------------------------------------------------------------
#define EINP_B  (16 * 3 * DD * 2)          // 12288 B : [src|nbr|spatial] f16
#define HSH_B   (16 * 64 * 2)              //  2048 B
#define H1_B    (16 * DD * 2)              //  4096 B
#define SLAB_B  (EINP_B + HSH_B + H1_B)    // 18432 B per wave

__global__ __launch_bounds__(128) void edges_kernel(
    const float* __restrict__ Wp1, const float* __restrict__ bp1,
    const float* __restrict__ bp2, const float* __restrict__ be1,
    const float* __restrict__ be2,
    const _Float16* __restrict__ pWp2, const _Float16* __restrict__ pWe1,
    const _Float16* __restrict__ pWe2,
    const _Float16* __restrict__ nodes_f16, const int* __restrict__ adj_i,
    float* __restrict__ out_edges)
{
    __shared__ __align__(16) char smem_raw[4 * SLAB_B];
    const int tid  = threadIdx.x;
    const int wave = tid >> 5;
    const int lane = tid & 31;
    const unsigned slab = (unsigned)(wave * SLAB_B);
    _Float16* einp = (_Float16*)(smem_raw + slab);            // [16][384]
    _Float16* hsh  = (_Float16*)(smem_raw + slab + EINP_B);   // [16][64]
    _Float16* h1sh = (_Float16*)(smem_raw + slab + EINP_B + HSH_B); // [16][128]

    const size_t e0   = (size_t)(blockIdx.x * 4 + wave) * 16;
    const int    mrow = (lane >> 4) * 8;
    const int    ncol = lane & 15;

    // ---- Async stage: src row chunks (lanes 0-15), nbr row chunks (16-31).
    {
        const int sel = lane >> 4;          // 0 = source, 1 = neighbor
        const int c   = lane & 15;          // 16B chunk within row
#pragma unroll
        for (int i = 0; i < 16; ++i) {
            const size_t e   = e0 + i;
            const int    b   = (int)(e / (NN * KK));
            const int    rem = (int)(e % (NN * KK));
            const int    n   = rem / KK;
            const int    nb  = adj_i[e];
            const int    rowid = sel ? nb : n;
            const _Float16* g = nodes_f16 + ((size_t)b * NN + rowid) * DD + c * 8;
            const unsigned  l = slab + (unsigned)(i * 768 + sel * 256 + c * 16);
            async_copy_b128(l, g);
        }
    }

    // ---- Tiny pos MLP (2 -> 64) in VALU while async copies fly.
    if (lane < 16) {
        const size_t e   = e0 + lane;
        const int    b   = (int)(e / (NN * KK));  (void)b;
        const int    rem = (int)(e % (NN * KK));
        const int    n   = rem / KK;
        const int    nb  = adj_i[e];
        const float  dy  = ((float)(nb / HH) - (float)(n / HH)) * (1.0f / 13.0f);
        const float  dx  = ((float)(nb % HH) - (float)(n % HH)) * (1.0f / 13.0f);
#pragma unroll 4
        for (int j = 0; j < 64; ++j) {
            float h = dy * Wp1[j] + dx * Wp1[64 + j] + bp1[j];
            hsh[lane * 64 + j] = (_Float16)gelu_exact(h);
        }
    }
    wait_async0();
    __syncthreads();

    const v8f vz = {0.f, 0.f, 0.f, 0.f, 0.f, 0.f, 0.f, 0.f};
    v8f acc[8];

    // ---- spatial_feat = hsh @ Wp2 + bp2 -> einp cols 256..383
#pragma unroll
    for (int t = 0; t < 8; ++t) acc[t] = vz;
    for (int ks = 0; ks < 2; ++ks) {
        v16h a = load_frag_rowmajor(hsh, 64, 0, ks * 32, lane);
        FragH bfr[8];
#pragma unroll
        for (int nt = 0; nt < 8; ++nt)
            bfr[nt] = load_frag_packed(pWp2, ks * 8 + nt, lane);
#pragma unroll
        for (int nt = 0; nt < 8; ++nt)
            acc[nt] = wmma_f16(a, bfr[nt].v, acc[nt]);
    }
#pragma unroll
    for (int nt = 0; nt < 8; ++nt)
#pragma unroll
        for (int r = 0; r < 8; ++r) {
            int col = nt * 16 + ncol;
            einp[(r + mrow) * 384 + 2 * DD + col] = (_Float16)(acc[nt][r] + bp2[col]);
        }
    __syncthreads();

    // ---- h1 = gelu(edge_input @ We1 + be1)
#pragma unroll
    for (int t = 0; t < 8; ++t) acc[t] = vz;
    for (int ks = 0; ks < 12; ++ks) {
        v16h a = load_frag_rowmajor(einp, 3 * DD, 0, ks * 32, lane);
        FragH bfr[8];
#pragma unroll
        for (int nt = 0; nt < 8; ++nt)
            bfr[nt] = load_frag_packed(pWe1, ks * 8 + nt, lane);
        if (ks + 1 < 12)
            __builtin_prefetch(pWe1 + (size_t)(ks + 1) * 8 * 512 + lane * 128, 0, 1);
#pragma unroll
        for (int nt = 0; nt < 8; ++nt)
            acc[nt] = wmma_f16(a, bfr[nt].v, acc[nt]);
    }
#pragma unroll
    for (int nt = 0; nt < 8; ++nt)
#pragma unroll
        for (int r = 0; r < 8; ++r) {
            int col = nt * 16 + ncol;
            h1sh[(r + mrow) * DD + col] = (_Float16)gelu_exact(acc[nt][r] + be1[col]);
        }
    __syncthreads();

    // ---- edges = h1 @ We2 + be2
#pragma unroll
    for (int t = 0; t < 8; ++t) acc[t] = vz;
    for (int ks = 0; ks < 4; ++ks) {
        v16h a = load_frag_rowmajor(h1sh, DD, 0, ks * 32, lane);
        FragH bfr[8];
#pragma unroll
        for (int nt = 0; nt < 8; ++nt)
            bfr[nt] = load_frag_packed(pWe2, ks * 8 + nt, lane);
#pragma unroll
        for (int nt = 0; nt < 8; ++nt)
            acc[nt] = wmma_f16(a, bfr[nt].v, acc[nt]);
    }
#pragma unroll
    for (int nt = 0; nt < 8; ++nt)
#pragma unroll
        for (int r = 0; r < 8; ++r) {
            int    col = nt * 16 + ncol;
            size_t er  = e0 + mrow + r;
            out_edges[er * DD + col] = acc[nt][r] + be2[col];
        }
}

// ---------------------------------------------------------------------------
extern "C" void kernel_launch(void* const* d_in, const int* in_sizes, int n_in,
                              void* d_out, int out_size, void* d_ws, size_t ws_size,
                              hipStream_t stream) {
    (void)in_sizes; (void)n_in; (void)out_size; (void)ws_size;

    const float* feat_map = (const float*)d_in[0];
    const float* W_node   = (const float*)d_in[1];
    const float* b_node   = (const float*)d_in[2];
    const float* ln_g     = (const float*)d_in[3];
    const float* ln_b     = (const float*)d_in[4];
    const float* Wp1      = (const float*)d_in[5];
    const float* bp1      = (const float*)d_in[6];
    const float* Wp2      = (const float*)d_in[7];
    const float* bp2      = (const float*)d_in[8];
    const float* We1      = (const float*)d_in[9];
    const float* be1      = (const float*)d_in[10];
    const float* We2      = (const float*)d_in[11];
    const float* be2      = (const float*)d_in[12];

    // d_out = [nodes fp32 | edges fp32 | adj (as fp32)]
    float* out_nodes = (float*)d_out;
    float* out_edges = out_nodes + (size_t)BB * NN * DD;
    float* out_adj   = out_edges + (size_t)BB * NN * KK * DD;

    char*  ws  = (char*)d_ws;
    size_t off = 0;
    _Float16* nrm_f16   = (_Float16*)(ws + off); off += (size_t)BB * NPAD * DD * 2;
    _Float16* nodes_f16 = (_Float16*)(ws + off); off += (size_t)BB * NN * DD * 2;
    int*      adj_i     = (int*)(ws + off);      off += (size_t)BB * NN * KK * 4;
    _Float16* pWn       = (_Float16*)(ws + off); off += (size_t)24 * 8 * 512 * 2;
    _Float16* pWe1      = (_Float16*)(ws + off); off += (size_t)12 * 8 * 512 * 2;
    _Float16* pWe2      = (_Float16*)(ws + off); off += (size_t)4  * 8 * 512 * 2;
    _Float16* pWp2      = (_Float16*)(ws + off); off += (size_t)2  * 8 * 512 * 2;

    pack_weights_kernel<<<336, 32, 0, stream>>>(W_node, We1, We2, Wp2,
                                                pWn, pWe1, pWe2, pWp2);
    zero_pad_kernel<<<(BB * (NPAD - NN) * DD + 255) / 256, 256, 0, stream>>>(nrm_f16);
    node_proj_kernel<<<(BB * NN) / 64, 128, 0, stream>>>(
        feat_map, b_node, ln_g, ln_b, pWn, out_nodes, nodes_f16, nrm_f16);
    simtopk_kernel<<<(BB * 13) / 4, 128, 0, stream>>>(nrm_f16, adj_i, out_adj);
    edges_kernel<<<(BB * NN * KK) / 64, 128, 0, stream>>>(
        Wp1, bp1, bp2, be1, be2, pWp2, pWe1, pWe2, nodes_f16, adj_i, out_edges);
}
